// BilstmCRFSequenceTagger1_50757923504758
// MI455X (gfx1250) — matrix operand order repaired and checked by reading, not compiled, not run on hardware
//
#include <hip/hip_runtime.h>
#include <hip/hip_bf16.h>

// ---------------------------------------------------------------------------
// BiLSTM-CRF tagger for MI455X (gfx1250, wave32, WMMA bf16 16x16x32).
// ---------------------------------------------------------------------------

typedef __attribute__((ext_vector_type(16))) __bf16 v16bf;
typedef __attribute__((ext_vector_type(8)))  float  v8f;

#define LOW_POT (-10000.0f)

__device__ __forceinline__ unsigned short f2bf(float f) {
    union { float f; unsigned u; } v; v.f = f;
    unsigned r = v.u + 0x7FFFu + ((v.u >> 16) & 1u);   // round-to-nearest-even
    return (unsigned short)(r >> 16);
}
__device__ __forceinline__ float bf2f(unsigned short h) {
    union { unsigned u; float f; } v; v.u = ((unsigned)h) << 16;
    return v.f;
}
__device__ __forceinline__ float sigm(float x) { return 1.0f / (1.0f + __expf(-x)); }

// A-operand (16-bit, 16x32) lane/slot mapping per ISA 7.12.2.
// element (m, kk) -> index into a 512-entry lane-major fragment [lane][16]
__device__ __forceinline__ int amap(int m, int kk) {
    int k16 = kk & 15, hi = kk >> 4;
    int lane = ((k16 >> 3) << 4) + m;          // lanes 0-15 / 16-31 halves
    int vgpr = hi * 4 + ((k16 & 7) >> 1);
    int pos  = vgpr * 2 + (kk & 1);
    return lane * 16 + pos;
}

__device__ __forceinline__ v8f wmma_bf16(const unsigned short* ap,
                                         const unsigned short* bp, v8f c) {
    v16bf a = *(const v16bf*)ap;               // 32B contiguous per lane
    v16bf b = *(const v16bf*)bp;
    return __builtin_amdgcn_wmma_f32_16x16x32_bf16(
        false, a, false, b, (short)0, c, false, false);
}

// ---------------------------------------------------------------------------
// Pack a row-major weight W[N][Kin] into B-operand fragments
// out[kt][nt][lane][pos], zero-padded to Kt*32 in K.
// B 32x16 layout: lanes 0-15 hold K=0..7,16..23, lanes 16-31 hold K=8..15,24..31; n=lane%16.
// ---------------------------------------------------------------------------
__global__ void k_packB(const float* __restrict__ W, int Kin, int Kt, int Nt,
                        unsigned short* __restrict__ out) {
    int total = Kt * Nt * 512;
    for (int e = blockIdx.x * blockDim.x + threadIdx.x; e < total;
         e += gridDim.x * blockDim.x) {
        int kt   = e / (Nt * 512);
        int rem  = e - kt * (Nt * 512);
        int nt   = rem >> 9;
        int lp   = rem & 511;
        int lane = lp >> 4, pos = lp & 15;
        int n = nt * 16 + (lane & 15);
        int k = kt * 32 + ((pos >> 3) << 4) + ((lane >> 4) << 3) + (pos & 7);
        float v = (k < Kin) ? W[(long)n * Kin + k] : 0.0f;
        out[e] = f2bf(v);
    }
}

// ---------------------------------------------------------------------------
// Input projection: xproj[dir][r][g] = emb[r] @ W_ih[dir]^T + b  (r = b*T+t)
// One block per (dir, rowtile). Embedding gather fused; staged as A-op in LDS.
// ---------------------------------------------------------------------------
__global__ __launch_bounds__(256) void k_xproj(
    const int* __restrict__ x, const float* __restrict__ embT,
    const unsigned short* __restrict__ WihOp,      // [2][10][64][512]
    const float* __restrict__ bF, const float* __restrict__ bB,
    float* __restrict__ xproj)                     // [2][16384][1024]
{
    __shared__ unsigned short aop[10 * 512];
    int blk = blockIdx.x;
    int dir = blk >> 10;
    int rt  = blk & 1023;
    int tid = threadIdx.x;

    for (int e = tid; e < 16 * 320; e += 256) {    // 20 per thread, uniform
        int m = e / 320, k = e - m * 320;
        float v = 0.0f;
        if (k < 300) {
            int xi = x[rt * 16 + m];
            v = embT[(long)xi * 300 + k];
        }
        aop[(k >> 5) * 512 + amap(m, k & 31)] = f2bf(v);
    }
    __syncthreads();

    int w = tid >> 5, lane = tid & 31;
    const float* bias = dir ? bB : bF;
    const unsigned short* wb = WihOp + (long)dir * 10 * 64 * 512;

    v8f acc[8];
    for (int i = 0; i < 8; i++) for (int q = 0; q < 8; q++) acc[i][q] = 0.0f;

    for (int kt = 0; kt < 10; kt++) {
        const unsigned short* ap = aop + kt * 512 + lane * 16;
        for (int i = 0; i < 8; i++) {
            int nt = w * 8 + i;
            acc[i] = wmma_bf16(ap, wb + ((kt * 64 + nt) << 9) + lane * 16, acc[i]);
        }
    }
    for (int i = 0; i < 8; i++) {
        int nt  = w * 8 + i;
        int col = nt * 16 + (lane & 15);
        int mb  = (lane >> 4) * 8;
        for (int q = 0; q < 8; q++) {
            int r = rt * 16 + mb + q;
            xproj[(((long)dir << 14) + r) * 1024 + col] = acc[i][q] + bias[col];
        }
    }
}

// ---------------------------------------------------------------------------
// LSTM recurrence. grid = 8 blocks: dir(2) x batch-chunk(4, 16 rows each).
// Per step: stage h -> bf16 A-op in LDS, 16 waves x 4 ntiles WMMA vs W_hh
// (operand-order, L2-resident), gates in fp32 with c-state in registers.
// ---------------------------------------------------------------------------
__global__ __launch_bounds__(512) void k_lstm(
    const unsigned short* __restrict__ WhhOp,      // [2][8][64][512]
    const float* __restrict__ xproj,               // [2][16384][1024]
    float* __restrict__ hbuf)                      // [2][64][256][256]
{
    __shared__ float hsh[16 * 256];                // 16 KB
    __shared__ unsigned short gsh[16 * 1024];      // 32 KB; first 8 KB aliased as aop
    unsigned short* aop = gsh;

    int dir  = blockIdx.x >> 2;
    int b0   = (blockIdx.x & 3) * 16;
    int tid  = threadIdx.x;
    int w    = tid >> 5, lane = tid & 31;
    int brow = tid >> 5;                           // 0..15 (batch row in tile)
    int j0   = (tid & 31) * 8;                     // hidden slice owned

    float creg[8];
    for (int q = 0; q < 8; q++) creg[q] = 0.0f;
    for (int e = tid; e < 16 * 256; e += 512) hsh[e] = 0.0f;
    __syncthreads();

    const unsigned short* wb = WhhOp + (long)dir * 8 * 64 * 512;

    for (int t = 0; t < 256; t++) {
        int tcur = dir ? (255 - t) : t;

        // 1) stage A operand (h tile, bf16) -- aliases low gsh
        for (int q = 0; q < 8; q++) {
            int j = j0 + q;
            aop[(j >> 5) * 512 + amap(brow, j & 31)] = f2bf(hsh[brow * 256 + j]);
        }
        __syncthreads();

        // 2) recurrent GEMM: [16 x 256] x [256 x 1024]
        v8f acc[4];
        for (int i = 0; i < 4; i++) for (int q = 0; q < 8; q++) acc[i][q] = 0.0f;
        for (int kt = 0; kt < 8; kt++) {
            const unsigned short* ap = aop + kt * 512 + lane * 16;
            for (int i = 0; i < 4; i++) {
                int nt = w * 4 + i;
                acc[i] = wmma_bf16(ap, wb + ((kt * 64 + nt) << 9) + lane * 16, acc[i]);
            }
        }
        __syncthreads();                            // all aop reads done

        // 3) publish recurrent preactivations (bf16) to gsh
        for (int i = 0; i < 4; i++) {
            int col = (w * 4 + i) * 16 + (lane & 15);
            int mb  = (lane >> 4) * 8;
            for (int q = 0; q < 8; q++)
                gsh[(mb + q) * 1024 + col] = f2bf(acc[i][q]);
        }
        __syncthreads();

        // 4) gates (fp32): c' = s(f)*c + s(i)*tanh(g); h' = s(o)*tanh(c')
        {
            long r = (long)dir * 16384 + (long)(b0 + brow) * 256 + tcur;
            const float* xp = xproj + r * 1024;
            for (int q = 0; q < 8; q++) {
                int j = j0 + q;
                float gi = xp[j]       + bf2f(gsh[brow * 1024 + j]);
                float gf = xp[256 + j] + bf2f(gsh[brow * 1024 + 256 + j]);
                float gg = xp[512 + j] + bf2f(gsh[brow * 1024 + 512 + j]);
                float go = xp[768 + j] + bf2f(gsh[brow * 1024 + 768 + j]);
                float c  = sigm(gf) * creg[q] + sigm(gi) * tanhf(gg);
                float h  = sigm(go) * tanhf(c);
                creg[q] = c;
                hsh[brow * 256 + j] = h;
                hbuf[(((long)dir * 64 + b0 + brow) * 256 + tcur) * 256 + j] = h;
            }
        }
        __syncthreads();
    }
}

// ---------------------------------------------------------------------------
// fc1 (WMMA) + relu + cls (VALU) + CRF padding. One block per rowtile.
// ---------------------------------------------------------------------------
__global__ __launch_bounds__(256) void k_fc1cls(
    const float* __restrict__ hbuf,                // [2][64][256][256]
    const unsigned short* __restrict__ fc1Op,      // [16][8][512]
    const float* __restrict__ fc1b, const float* __restrict__ clsW,
    const float* __restrict__ clsb, float* __restrict__ unary)  // [16384][19]
{
    __shared__ unsigned short aop[16 * 512];       // 16 KB
    __shared__ float f1[16 * 128];                 // 8 KB
    int rt = blockIdx.x, tid = threadIdx.x;

    for (int e = tid; e < 16 * 512; e += 256) {    // 32 per thread, uniform
        int m = e >> 9, k = e & 511;
        int r = rt * 16 + m, b = r >> 8, t = r & 255;
        int dirc = k >> 8, kh = k & 255;
        float v = hbuf[(((long)dirc * 64 + b) * 256 + t) * 256 + kh];
        aop[(k >> 5) * 512 + amap(m, k & 31)] = f2bf(v);
    }
    __syncthreads();

    int w = tid >> 5, lane = tid & 31;
    v8f acc;
    for (int q = 0; q < 8; q++) acc[q] = 0.0f;
    for (int kt = 0; kt < 16; kt++)
        acc = wmma_bf16(aop + kt * 512 + lane * 16,
                        fc1Op + ((kt * 8 + w) << 9) + lane * 16, acc);
    {
        int col = w * 16 + (lane & 15);
        int mb  = (lane >> 4) * 8;
        for (int q = 0; q < 8; q++) {
            float z = acc[q] + fc1b[col];
            f1[(mb + q) * 128 + col] = z > 0.0f ? z : 0.0f;  // relu
        }
    }
    __syncthreads();

    if (tid < 272) {                                // 16 rows x 17 labels
        int m = tid / 17, lab = tid - m * 17;
        float s = clsb[lab];
        const float* wr = clsW + lab * 128;
        const float* fr = f1 + m * 128;
        for (int k = 0; k < 128; k++) s += fr[k] * wr[k];
        unary[((long)(rt * 16 + m)) * 19 + lab] = s;
    } else if (tid < 304) {                         // START/END padding
        int e = tid - 272, m = e >> 1, lab = 17 + (e & 1);
        unary[((long)(rt * 16 + m)) * 19 + lab] = LOW_POT;
    }
}

// ---------------------------------------------------------------------------
// Viterbi: one wave per batch element. trans_pn[prev][next] = trans[next][prev].
// argmax uses strict '>' to match JAX first-occurrence semantics.
// ---------------------------------------------------------------------------
__global__ __launch_bounds__(32) void k_viterbi(
    const float* __restrict__ unary, const float* __restrict__ trans,
    float* __restrict__ out)
{
    __shared__ float tr[361], sc[19], nsc[19];
    __shared__ unsigned char bp[256 * 19];
    int b = blockIdx.x, lane = threadIdx.x;

    for (int e = lane; e < 361; e += 32) tr[e] = trans[e];
    if (lane < 19) sc[lane] = (lane == 17) ? 0.0f : LOW_POT;  // START
    __syncthreads();

    for (int t = 0; t < 256; t++) {
        if (lane < 19) {
            float best = sc[0] + tr[lane * 19 + 0];
            int bpv = 0;
            for (int p = 1; p < 19; p++) {
                float v = sc[p] + tr[lane * 19 + p];
                if (v > best) { best = v; bpv = p; }
            }
            nsc[lane] = best + unary[((long)b * 256 + t) * 19 + lane];
            bp[t * 19 + lane] = (unsigned char)bpv;
        }
        __syncthreads();
        if (lane < 19) sc[lane] = nsc[lane];
        __syncthreads();
    }
    if (lane < 19) sc[lane] += tr[18 * 19 + lane];  // + trans[END][l]
    __syncthreads();

    if (lane == 0) {
        float best = sc[0]; int last = 0;
        for (int l = 1; l < 19; l++) if (sc[l] > best) { best = sc[l]; last = l; }
        out[b] = best;
        int cur = last;
        for (int t = 255; t >= 0; t--) {            // labels[t]=cur; cur=bp[t][cur]
            out[64 + (long)b * 256 + t] = (float)cur;
            cur = bp[t * 19 + cur];
        }
    }
}

// ---------------------------------------------------------------------------
extern "C" void kernel_launch(void* const* d_in, const int* in_sizes, int n_in,
                              void* d_out, int out_size, void* d_ws, size_t ws_size,
                              hipStream_t stream) {
    const int*   x     = (const int*)  d_in[0];
    const float* embT  = (const float*)d_in[1];
    const float* WihF  = (const float*)d_in[2];
    const float* WhhF  = (const float*)d_in[3];
    const float* bF    = (const float*)d_in[4];
    const float* WihB  = (const float*)d_in[5];
    const float* WhhB  = (const float*)d_in[6];
    const float* bB    = (const float*)d_in[7];
    const float* fc1W  = (const float*)d_in[8];
    const float* fc1b  = (const float*)d_in[9];
    const float* clsW  = (const float*)d_in[10];
    const float* clsb  = (const float*)d_in[11];
    const float* trans = (const float*)d_in[12];

    // Workspace layout (all offsets 256B-aligned); total ~165 MB
    char* ws = (char*)d_ws;
    unsigned short* WihOp = (unsigned short*)(ws);                 // 1,310,720 B
    unsigned short* WhhOp = (unsigned short*)(ws + 1310720);       // 1,048,576 B
    unsigned short* fc1Op = (unsigned short*)(ws + 2359296);       //   131,072 B
    float* xproj = (float*)(ws + 2490368);                         // 134,217,728 B
    float* hbuf  = (float*)(ws + 136708096);                       //  33,554,432 B
    float* unary = (float*)(ws + 170262528);                       //   1,245,184 B

    // 1) weight packing into WMMA B-operand fragment order
    k_packB<<<320, 256, 0, stream>>>(WihF, 300, 10, 64, WihOp);
    k_packB<<<320, 256, 0, stream>>>(WihB, 300, 10, 64, WihOp + 10 * 64 * 512);
    k_packB<<<256, 256, 0, stream>>>(WhhF, 256,  8, 64, WhhOp);
    k_packB<<<256, 256, 0, stream>>>(WhhB, 256,  8, 64, WhhOp + 8 * 64 * 512);
    k_packB<<< 64, 256, 0, stream>>>(fc1W, 512, 16,  8, fc1Op);

    // 2) embedding gather + input projection GEMM (both directions)
    k_xproj<<<2048, 256, 0, stream>>>(x, embT, WihOp, bF, bB, xproj);

    // 3) bidirectional LSTM recurrence (8 resident workgroups)
    k_lstm<<<8, 512, 0, stream>>>(WhhOp, xproj, hbuf);

    // 4) fc1 (WMMA) + relu + classifier + CRF padding
    k_fc1cls<<<1024, 256, 0, stream>>>(hbuf, fc1Op, fc1b, clsW, clsb, unary);

    // 5) Viterbi decode -> d_out = [path_score(64) | labels(64*256) as float]
    k_viterbi<<<64, 32, 0, stream>>>(unary, trans, (float*)d_out);
}
